// LinearTransformerEncoderLayer_69913477644663
// MI455X (gfx1250) — matrix operand order repaired and checked by reading, not compiled, not run on hardware
//
#include <hip/hip_runtime.h>

// ---------------------------------------------------------------------------
// LinearTransformerEncoderLayer for MI455X (gfx1250, wave32, WMMA).
// GEMMs: v_wmma_f32_16x16x32_bf16 fed from LDS, tiles staged with CDNA5
// async global->LDS copies (ASYNCcnt), double buffered.
// Everything else (LN, ELU feature map, kv/z reduction, GELU, residuals)
// is exact fp32.
// ---------------------------------------------------------------------------

typedef __attribute__((ext_vector_type(16))) __bf16 v16bf;
typedef __attribute__((ext_vector_type(8)))  __bf16 v8bf;
typedef __attribute__((ext_vector_type(8)))  float  v8f;

#define D_MODEL 1024
#define D_FF    4096
#define NTOK    16384   // 4 * 4096
#define NHEAD   16
#define HEADD   64

__device__ __forceinline__ unsigned short f2bf_u(float f) {
  unsigned u = __float_as_uint(f);
  unsigned r = u + 0x7fffu + ((u >> 16) & 1u);   // round-to-nearest-even
  return (unsigned short)(r >> 16);
}

__device__ __forceinline__ float gelu_exact(float x) {
  return 0.5f * x * (1.0f + erff(x * 0.70710678118654752f));
}

// phi(x) = elu(x) + 1 = x+1 (x>0) else exp(x)
__device__ __forceinline__ float phi(float x) {
  return (x > 0.0f) ? (x + 1.0f) : __expf(x);
}

// ------------------ CDNA5 async global -> LDS copy helpers ------------------
// global_load_async_to_lds_b128: VDST = VGPR holding LDS byte address,
// VADDR = 64-bit global address.  Tracked with ASYNCcnt.
__device__ __forceinline__ void async_ld16(unsigned lds_off,
                                           const unsigned short* g) {
  asm volatile("global_load_async_to_lds_b128 %0, %1, off"
               :: "v"(lds_off), "v"(g)
               : "memory");
}
__device__ __forceinline__ void wait_async0() {
  asm volatile("s_wait_asynccnt 0" ::: "memory");
}
// Low 32 bits of a generic pointer to LDS == LDS byte offset (flat aperture
// mapping truncates the upper bits).
__device__ __forceinline__ unsigned lds_addr32(const void* p) {
  return (unsigned)(unsigned long long)p;
}

// --------------------------- fp32 -> bf16 convert ---------------------------
__global__ __launch_bounds__(256) void k_cvt(const float* __restrict__ in,
                                             unsigned short* __restrict__ out,
                                             int n) {
  int i = blockIdx.x * 256 + threadIdx.x;
  if (i < n) out[i] = f2bf_u(in[i]);
}

// --------------------------- LayerNorm (C = 1024) ---------------------------
__global__ __launch_bounds__(256) void k_ln(const float* __restrict__ x,
                                            const float* __restrict__ g,
                                            const float* __restrict__ b,
                                            unsigned short* __restrict__ out) {
  const int row = blockIdx.x;
  const int t = threadIdx.x;
  const float* xr = x + (size_t)row * D_MODEL;
  float4 v = ((const float4*)xr)[t];
  float s  = v.x + v.y + v.z + v.w;
  float ss = v.x * v.x + v.y * v.y + v.z * v.z + v.w * v.w;
  __shared__ float rs[256];
  __shared__ float rq[256];
  rs[t] = s; rq[t] = ss;
  __syncthreads();
  for (int o = 128; o > 0; o >>= 1) {
    if (t < o) { rs[t] += rs[t + o]; rq[t] += rq[t + o]; }
    __syncthreads();
  }
  float mu  = rs[0] * (1.0f / D_MODEL);
  float var = rq[0] * (1.0f / D_MODEL) - mu * mu;
  float inv = rsqrtf(var + 1e-5f);
  int c = t * 4;
  float4 gv = ((const float4*)g)[t];
  float4 bv = ((const float4*)b)[t];
  unsigned short* orow = out + (size_t)row * D_MODEL + c;
  orow[0] = f2bf_u((v.x - mu) * inv * gv.x + bv.x);
  orow[1] = f2bf_u((v.y - mu) * inv * gv.y + bv.y);
  orow[2] = f2bf_u((v.z - mu) * inv * gv.z + bv.z);
  orow[3] = f2bf_u((v.w - mu) * inv * gv.w + bv.w);
}

// ------------------------------- WMMA GEMM ----------------------------------
// C[M,N] = A_bf16[M,K] @ B_bf16[K,N] + bias.
//   EPI=0: store fp32    EPI=1: store fp32 + residual    EPI=2: GELU -> bf16
//
// Block tile 128(M) x 64(N), 256 threads = 8 waves; wave w owns rows
// [bm+16w, bm+16w+16) x all 64 N (4 WMMA accumulators -> one A fragment is
// reused by 4 WMMAs).  K stepped by 32.  A and B K-slices are staged in LDS
// by async global->LDS b128 copies, double buffered; LDS rows are padded
// (A: 80B, B: 144B) so per-lane ds_load_b128 fragment reads are spread
// across bank groups.
//
// ISA fragment layouts (CDNA5 ISA), half = lane>>4:
//   A 16x32 bf16: lane holds row M=lane%16; elems 0..7 = K 8h+0..7,
//                 elems 8..15 = K 16+8h+0..7  (two 16B chunks).
//   B 32x16 bf16: lane holds col N=lane%16; elems 0..15 = same K order as A.
//   C/D f32:      lane holds col N=lane%16; VGPR r = row r + 8h.
#define GA_STRIDE 80    // bytes per A row in LDS (32 bf16 + pad)
#define GB_STRIDE 144   // bytes per B row in LDS (64 bf16 + pad)
#define GA_BUF    (128 * GA_STRIDE)            // 10240 B
#define GB_BUF    (32 * GB_STRIDE)             // 4608 B
#define GS_TOTAL  (2 * GA_BUF + 2 * GB_BUF)    // 29696 B

template <int EPI>
__global__ __launch_bounds__(256)
void k_gemm(const unsigned short* __restrict__ A,
            const unsigned short* __restrict__ B,
            const float* __restrict__ bias,
            const float* __restrict__ resid,
            float* __restrict__ Cf,
            unsigned short* __restrict__ Cb,
            int M, int N, int K) {
  __shared__ __align__(16) char smem[GS_TOTAL];
  const int t    = threadIdx.x;
  const int lane = t & 31;
  const int wave = t >> 5;          // 0..7 : M sub-tile
  const int bm   = blockIdx.y << 7; // 128-row block tile
  const int bn   = blockIdx.x << 6; // 64-col block tile
  const int half = lane >> 4;
  const unsigned smbase = lds_addr32(smem);

  // per-thread copy assignments (one K-step = A 512 chunks + B 256 chunks)
  const int ar0 = t >> 2,          ac0 = t & 3;        // A chunk 1
  const int ar1 = (t + 256) >> 2,  ac1 = t & 3;        // A chunk 2
  const int br  = t >> 3,          bc  = t & 7;        // B chunk

  auto stage = [&](int buf, int k0) {
    unsigned aoff = smbase + buf * GA_BUF;
    unsigned boff = smbase + 2 * GA_BUF + buf * GB_BUF;
    async_ld16(aoff + ar0 * GA_STRIDE + ac0 * 16,
               A + (size_t)(bm + ar0) * K + k0 + ac0 * 8);
    async_ld16(aoff + ar1 * GA_STRIDE + ac1 * 16,
               A + (size_t)(bm + ar1) * K + k0 + ac1 * 8);
    async_ld16(boff + br * GB_STRIDE + bc * 16,
               B + (size_t)(k0 + br) * N + bn + bc * 8);
  };

  v8f c0 = {}, c1 = {}, c2 = {}, c3 = {};
  const int nk = K >> 5;
  stage(0, 0);
  for (int kt = 0; kt < nk; ++kt) {
    const int cur = kt & 1;
    wait_async0();
    __syncthreads();
    if (kt + 1 < nk) stage(cur ^ 1, (kt + 1) << 5);

    const char* ab = smem + cur * GA_BUF +
                     (wave * 16 + (lane & 15)) * GA_STRIDE + half * 16;
    v8bf alo = *(const v8bf*)(ab);        // K = 8h + 0..7
    v8bf ahi = *(const v8bf*)(ab + 32);   // K = 16 + 8h + 0..7
    v16bf a;
#pragma unroll
    for (int e = 0; e < 8; ++e) { a[e] = alo[e]; a[e + 8] = ahi[e]; }

    const char* bb = smem + 2 * GA_BUF + cur * GB_BUF + (lane & 15) * 2;
    v16bf b0, b1, b2, b3;
#pragma unroll
    for (int e = 0; e < 16; ++e) {
      int kr = (e < 8) ? (8 * half + e) : (16 + 8 * half + (e - 8));
      const __bf16* p = (const __bf16*)(bb + kr * GB_STRIDE);
      b0[e] = p[0]; b1[e] = p[16]; b2[e] = p[32]; b3[e] = p[48];
    }
    c0 = __builtin_amdgcn_wmma_f32_16x16x32_bf16(false, a, false, b0,
                                                 (short)0, c0, false, false);
    c1 = __builtin_amdgcn_wmma_f32_16x16x32_bf16(false, a, false, b1,
                                                 (short)0, c1, false, false);
    c2 = __builtin_amdgcn_wmma_f32_16x16x32_bf16(false, a, false, b2,
                                                 (short)0, c2, false, false);
    c3 = __builtin_amdgcn_wmma_f32_16x16x32_bf16(false, a, false, b3,
                                                 (short)0, c3, false, false);
    __syncthreads();
  }

  const int coln = bn + (lane & 15);
  const float bv0 = bias[coln];
  const float bv1 = bias[coln + 16];
  const float bv2 = bias[coln + 32];
  const float bv3 = bias[coln + 48];
#pragma unroll
  for (int r = 0; r < 8; ++r) {
    int mrow = bm + wave * 16 + 8 * half + r;
    size_t i0 = (size_t)mrow * N + coln;
    float v0 = c0[r] + bv0;
    float v1 = c1[r] + bv1;
    float v2 = c2[r] + bv2;
    float v3 = c3[r] + bv3;
    if (EPI == 0) {
      Cf[i0] = v0; Cf[i0 + 16] = v1; Cf[i0 + 32] = v2; Cf[i0 + 48] = v3;
    } else if (EPI == 1) {
      Cf[i0]      = v0 + resid[i0];
      Cf[i0 + 16] = v1 + resid[i0 + 16];
      Cf[i0 + 32] = v2 + resid[i0 + 32];
      Cf[i0 + 48] = v3 + resid[i0 + 48];
    } else {
      Cb[i0]      = f2bf_u(gelu_exact(v0));
      Cb[i0 + 16] = f2bf_u(gelu_exact(v1));
      Cb[i0 + 32] = f2bf_u(gelu_exact(v2));
      Cb[i0 + 48] = f2bf_u(gelu_exact(v3));
    }
  }
}

// ----------------- linear attention: kv = phi(k)^T v, z = sum phi(k) --------
__global__ __launch_bounds__(256) void k_kv(const float* __restrict__ qkv,
                                            float* __restrict__ kv,
                                            float* __restrict__ z) {
  const int bh = blockIdx.x;          // b*16 + h
  const int b = bh >> 4, h = bh & 15;
  const float* base = qkv + ((size_t)b * 4096) * (3 * D_MODEL) + h * HEADD;
  __shared__ float skf[4][HEADD];
  __shared__ float sv[4][HEADD];
  const int t = threadIdx.x;
  const int ty = t >> 4, tx = t & 15;  // 16x16 thread grid
  const int d0 = ty * 4, e0 = tx * 4;
  float acc[4][4] = {};
  float zacc[4] = {0.f, 0.f, 0.f, 0.f};

  for (int n0 = 0; n0 < 4096; n0 += 4) {
    __syncthreads();
    for (int i = t; i < 4 * HEADD; i += 256) {
      int j = i >> 6, d = i & 63;
      const float* rp = base + (size_t)(n0 + j) * (3 * D_MODEL);
      skf[j][d] = phi(rp[D_MODEL + d]);   // k slice
      sv[j][d]  = rp[2 * D_MODEL + d];    // v slice
    }
    __syncthreads();
#pragma unroll
    for (int j = 0; j < 4; ++j) {
      float kd[4], ve[4];
#pragma unroll
      for (int dd = 0; dd < 4; ++dd) kd[dd] = skf[j][d0 + dd];
#pragma unroll
      for (int ee = 0; ee < 4; ++ee) ve[ee] = sv[j][e0 + ee];
#pragma unroll
      for (int dd = 0; dd < 4; ++dd)
#pragma unroll
        for (int ee = 0; ee < 4; ++ee) acc[dd][ee] += kd[dd] * ve[ee];
      if (tx == 0) {
#pragma unroll
        for (int dd = 0; dd < 4; ++dd) zacc[dd] += kd[dd];
      }
    }
  }

  float* kvp = kv + (size_t)bh * HEADD * HEADD;
#pragma unroll
  for (int dd = 0; dd < 4; ++dd)
#pragma unroll
    for (int ee = 0; ee < 4; ++ee)
      kvp[(d0 + dd) * HEADD + (e0 + ee)] = acc[dd][ee];
  if (tx == 0) {
#pragma unroll
    for (int dd = 0; dd < 4; ++dd) z[(size_t)bh * HEADD + d0 + dd] = zacc[dd];
  }
}

// ----------------- y = (phi(q) @ kv) / max(phi(q) . z, eps), bf16 out -------
__global__ __launch_bounds__(256) void k_attnout(const float* __restrict__ qkv,
                                                 const float* __restrict__ kv,
                                                 const float* __restrict__ z,
                                                 unsigned short* __restrict__ y) {
  const int tok = blockIdx.x;       // 0..16383
  const int b = tok >> 12;          // 4096 tokens per batch
  const int t = threadIdx.x;
  __shared__ float qf[D_MODEL];
  __shared__ float den[NHEAD];
  const float* qrow = qkv + (size_t)tok * (3 * D_MODEL);
  for (int i = t; i < D_MODEL; i += 256) qf[i] = phi(qrow[i]);
  __syncthreads();
  if (t < NHEAD) {
    const float* zp = z + ((size_t)b * NHEAD + t) * HEADD;
    float s = 0.f;
    for (int d = 0; d < HEADD; ++d) s += qf[t * HEADD + d] * zp[d];
    den[t] = fmaxf(s, 1e-6f);
  }
  __syncthreads();
  const int c0 = t * 4;             // 4 outputs/thread, same head
  const int h = c0 >> 6, e0 = c0 & 63;
  const float* kvp = kv + ((size_t)b * NHEAD + h) * (HEADD * HEADD);
  float num0 = 0.f, num1 = 0.f, num2 = 0.f, num3 = 0.f;
  for (int d = 0; d < HEADD; ++d) {
    float q = qf[h * HEADD + d];
    float4 kvv = *(const float4*)(kvp + d * HEADD + e0);
    num0 += q * kvv.x; num1 += q * kvv.y; num2 += q * kvv.z; num3 += q * kvv.w;
  }
  float r = 1.0f / den[h];
  unsigned short* yp = y + (size_t)tok * D_MODEL + c0;
  yp[0] = f2bf_u(num0 * r);
  yp[1] = f2bf_u(num1 * r);
  yp[2] = f2bf_u(num2 * r);
  yp[3] = f2bf_u(num3 * r);
}

// ------------------------------ orchestration --------------------------------
extern "C" void kernel_launch(void* const* d_in, const int* in_sizes, int n_in,
                              void* d_out, int out_size, void* d_ws, size_t ws_size,
                              hipStream_t stream) {
  const float* src   = (const float*)d_in[0];
  const float* qkv_w = (const float*)d_in[1];
  const float* qkv_b = (const float*)d_in[2];
  const float* out_w = (const float*)d_in[3];
  const float* out_b = (const float*)d_in[4];
  const float* w1    = (const float*)d_in[5];
  const float* b1    = (const float*)d_in[6];
  const float* w2    = (const float*)d_in[7];
  const float* b2    = (const float*)d_in[8];
  const float* g1    = (const float*)d_in[9];
  const float* be1   = (const float*)d_in[10];
  const float* g2    = (const float*)d_in[11];
  const float* be2   = (const float*)d_in[12];
  (void)in_sizes; (void)n_in; (void)out_size; (void)ws_size;

  char* wsp = (char*)d_ws;
  size_t off = 0;
  auto alloc = [&](size_t bytes) {
    size_t o = off;
    off += (bytes + 255) & ~(size_t)255;
    return o;
  };
  unsigned short* wqb  = (unsigned short*)(wsp + alloc((size_t)D_MODEL * 3 * D_MODEL * 2));
  unsigned short* wob  = (unsigned short*)(wsp + alloc((size_t)D_MODEL * D_MODEL * 2));
  unsigned short* w1b  = (unsigned short*)(wsp + alloc((size_t)D_MODEL * D_FF * 2));
  unsigned short* w2b  = (unsigned short*)(wsp + alloc((size_t)D_FF * D_MODEL * 2));
  unsigned short* xln  = (unsigned short*)(wsp + alloc((size_t)NTOK * D_MODEL * 2));
  float*          qkvf = (float*)(wsp + alloc((size_t)NTOK * 3 * D_MODEL * 4));
  float*          kvb  = (float*)(wsp + alloc((size_t)64 * HEADD * HEADD * 4));
  float*          zb   = (float*)(wsp + alloc((size_t)64 * HEADD * 4));
  unsigned short* ybf  = (unsigned short*)(wsp + alloc((size_t)NTOK * D_MODEL * 2));
  float*          xres = (float*)(wsp + alloc((size_t)NTOK * D_MODEL * 4));
  unsigned short* x2ln = (unsigned short*)(wsp + alloc((size_t)NTOK * D_MODEL * 2));
  unsigned short* hbf  = (unsigned short*)(wsp + alloc((size_t)NTOK * D_FF * 2));

  // 1) weights fp32 -> bf16
  {
    int n = D_MODEL * 3 * D_MODEL;
    k_cvt<<<(n + 255) / 256, 256, 0, stream>>>(qkv_w, wqb, n);
    n = D_MODEL * D_MODEL;
    k_cvt<<<(n + 255) / 256, 256, 0, stream>>>(out_w, wob, n);
    n = D_MODEL * D_FF;
    k_cvt<<<(n + 255) / 256, 256, 0, stream>>>(w1, w1b, n);
    n = D_FF * D_MODEL;
    k_cvt<<<(n + 255) / 256, 256, 0, stream>>>(w2, w2b, n);
  }

  // 2) LN1 -> bf16 activations
  k_ln<<<NTOK, 256, 0, stream>>>(src, g1, be1, xln);

  // 3) QKV = xln @ qkv_w + qkv_b (fp32 out)
  {
    dim3 g(3 * D_MODEL / 64, NTOK / 128);
    k_gemm<0><<<g, 256, 0, stream>>>(xln, wqb, qkv_b, nullptr, qkvf, nullptr,
                                     NTOK, 3 * D_MODEL, D_MODEL);
  }

  // 4) linear attention core
  k_kv<<<64, 256, 0, stream>>>(qkvf, kvb, zb);
  k_attnout<<<NTOK, 256, 0, stream>>>(qkvf, kvb, zb, ybf);

  // 5) x = src + y @ out_w + out_b
  {
    dim3 g(D_MODEL / 64, NTOK / 128);
    k_gemm<1><<<g, 256, 0, stream>>>(ybf, wob, out_b, src, xres, nullptr,
                                     NTOK, D_MODEL, D_MODEL);
  }

  // 6) LN2 -> bf16
  k_ln<<<NTOK, 256, 0, stream>>>(xres, g2, be2, x2ln);

  // 7) h = gelu(x2ln @ w1 + b1) (bf16 out)
  {
    dim3 g(D_FF / 64, NTOK / 128);
    k_gemm<2><<<g, 256, 0, stream>>>(x2ln, w1b, b1, nullptr, nullptr, hbf,
                                     NTOK, D_FF, D_MODEL);
  }

  // 8) out = x + h @ w2 + b2 (fp32 out)
  {
    dim3 g(D_MODEL / 64, NTOK / 128);
    k_gemm<1><<<g, 256, 0, stream>>>(hbf, w2b, b2, xres, (float*)d_out, nullptr,
                                     NTOK, D_MODEL, D_FF);
  }
}
